// CausalSelfPoM_3289944948861
// MI455X (gfx1250) — matrix operand order repaired and checked by reading, not compiled, not run on hardware
//
#include <hip/hip_runtime.h>
#include <stdint.h>

typedef __bf16 bf16_t;
typedef __attribute__((ext_vector_type(16))) __bf16 v16bf;
typedef __attribute__((ext_vector_type(8)))  __bf16 v8bf;
typedef __attribute__((ext_vector_type(8)))  float  v8f;
typedef __attribute__((ext_vector_type(4)))  unsigned int u32x4;
typedef __attribute__((ext_vector_type(8)))  int i32x8;
typedef __attribute__((ext_vector_type(4)))  int i32x4;

#define B_  4
#define T_  4096
#define C_  1024
#define H_  2048
#define M_  (B_*T_)      /* 16384 */
#define NC_ 32           /* scan chunks */
#define CT_ (T_/NC_)     /* 128 rows per chunk */

__device__ __forceinline__ bf16_t f2bf(float f) {
  union { float f; uint32_t u; } v; v.f = f;
  uint32_t r = v.u + 0x7FFFu + ((v.u >> 16) & 1u);   // round-to-nearest-even
  union { uint16_t u; bf16_t b; } o; o.u = (uint16_t)(r >> 16);
  return o.b;
}

// LDS byte offset of a __shared__ object: low 32 bits of the generic address
// are the LDS offset per the aperture mapping (ISA 10.2).
__device__ __forceinline__ uint32_t lds_off(const void* p) {
  return (uint32_t)(uintptr_t)p;
}

// ---- Tensor Data Mover descriptors (ISA 8.3/8.4), data_size = 2 bytes ----
// This toolchain exposes the 6-arg builtin:
//   (uint32x4 g0, int32x8 g1, int32x4 g2, int32x4 g3, int32x8 g4, i32 cpol)

// 2-D tile load: tile t0 x t1 elements from a [dim1][dim0] tensor (row stride s0).
__device__ __forceinline__ void tdm_load_2d(uint32_t lds, uint64_t gaddr,
    uint32_t dim0, uint32_t dim1, uint32_t t0, uint32_t t1, uint64_t s0) {
  u32x4 g0; i32x8 g1;
  i32x4 g2 = {0, 0, 0, 0}, g3 = {0, 0, 0, 0};
  i32x8 g4 = {0, 0, 0, 0, 0, 0, 0, 0};
  g0[0] = 1u;                                            // count=1, no gather
  g0[1] = lds;                                           // lds_addr
  g0[2] = (uint32_t)gaddr;                               // global_addr[31:0]
  g0[3] = ((uint32_t)(gaddr >> 32) & 0x01FFFFFFu) | (2u << 30);  // addr[56:32] | type=2
  g1[0] = (int)(1u << 16);                               // data_size = 2B
  g1[1] = (int)((dim0 & 0xFFFFu) << 16);                 // tensor_dim0 lo
  g1[2] = (int)((dim0 >> 16) | ((dim1 & 0xFFFFu) << 16));
  g1[3] = (int)((dim1 >> 16) | (t0 << 16));              // tile_dim0
  g1[4] = (int)t1;                                       // tile_dim1, tile_dim2=0 (2-D)
  g1[5] = (int)(uint32_t)s0;                             // tensor_dim0_stride lo
  g1[6] = (int)((uint32_t)(s0 >> 32) & 0xFFFFu);
  g1[7] = 0;
  __builtin_amdgcn_tensor_load_to_lds(g0, g1, g2, g3, g4, 0);
}

// 3-D tile load: t0 x t1 x t2 from [dim2][dim1][dim0] tensor, strides s0 (y), s1 (z).
__device__ __forceinline__ void tdm_load_3d(uint32_t lds, uint64_t gaddr,
    uint32_t dim0, uint32_t dim1, uint32_t dim2,
    uint32_t t0, uint32_t t1, uint32_t t2, uint64_t s0, uint64_t s1) {
  u32x4 g0; i32x8 g1;
  i32x4 g2, g3 = {0, 0, 0, 0};
  i32x8 g4 = {0, 0, 0, 0, 0, 0, 0, 0};
  g0[0] = 1u;
  g0[1] = lds;
  g0[2] = (uint32_t)gaddr;
  g0[3] = ((uint32_t)(gaddr >> 32) & 0x01FFFFFFu) | (2u << 30);
  g1[0] = (int)(1u << 16);
  g1[1] = (int)((dim0 & 0xFFFFu) << 16);
  g1[2] = (int)((dim0 >> 16) | ((dim1 & 0xFFFFu) << 16));
  g1[3] = (int)((dim1 >> 16) | (t0 << 16));
  g1[4] = (int)(t1 | (t2 << 16));
  g1[5] = (int)(uint32_t)s0;
  g1[6] = (int)(((uint32_t)(s0 >> 32) & 0xFFFFu) | (((uint32_t)s1 & 0xFFFFu) << 16));
  g1[7] = (int)(uint32_t)(s1 >> 16);
  g2[0] = (int)dim2; g2[1] = 0; g2[2] = 0; g2[3] = 0;    // tensor_dim2
  __builtin_amdgcn_tensor_load_to_lds(g0, g1, g2, g3, g4, 0);
}

// ---------------- Phase 0: conversions ----------------

__global__ void k_f32_to_bf16(const float* __restrict__ in, bf16_t* __restrict__ out, int n) {
  int i = blockIdx.x * blockDim.x + threadIdx.x;
  int stride = gridDim.x * blockDim.x;
  for (; i < n; i += stride) out[i] = f2bf(in[i]);
}

// in: K x N fp32 row-major; out: N x K bf16 row-major; out[n][k] = in[k][n]*scale[n]
__global__ void k_transpose_convert(const float* __restrict__ in,
                                    const float* __restrict__ scale,
                                    bf16_t* __restrict__ out, int K, int N) {
  __shared__ float tile[32][33];
  int k0 = blockIdx.x * 32, n0 = blockIdx.y * 32;
  for (int i = threadIdx.y; i < 32; i += 8)
    tile[i][threadIdx.x] = in[(size_t)(k0 + i) * N + (n0 + threadIdx.x)];
  __syncthreads();
  for (int i = threadIdx.y; i < 32; i += 8) {
    int n = n0 + i, k = k0 + threadIdx.x;
    float sc = scale ? scale[n] : 1.0f;
    out[(size_t)n * K + k] = f2bf(tile[threadIdx.x][i] * sc);
  }
}

// ---------------- Phase 1: fused sel/agg WMMA GEMM (TDM double-buffered) ----
// Block: 256 threads (8 waves as 4x2), BM=128, BN=64 per region, BK=32, 3 regions.
// Weight regions (sel, agg-chunk0, agg-chunk1) are contiguous: wcat[3][H][C].

__global__ __launch_bounds__(256)
void k_gemm_sel_agg(const bf16_t* __restrict__ xb,     // [M][C]
                    const bf16_t* __restrict__ wcat,   // [3][H][C] (pre-scaled)
                    float* __restrict__ s_out,         // [M][H]
                    float* __restrict__ poly_out) {    // [M][H]
  __shared__ __align__(16) bf16_t As[2][128][32];
  __shared__ __align__(16) bf16_t Bs[2][3][64][32];

  const int tid  = threadIdx.x;
  const int lane = tid & 31;
  const int wid  = tid >> 5;         // wave32: 8 waves
  const int wm   = wid >> 1;         // 0..3 -> 32-row strip
  const int wn   = wid & 1;          // 0..1 -> 32-col strip
  const int m0   = blockIdx.x * 128;
  const int n0   = blockIdx.y * 64;

  v8f acc[3][2][2] = {};
  const int fr = lane & 15, kh = lane >> 4;

  const uint64_t aBase = (uint64_t)(uintptr_t)xb   + (uint64_t)m0 * C_ * 2;
  const uint64_t bBase = (uint64_t)(uintptr_t)wcat + (uint64_t)n0 * C_ * 2;
  const int KSTEPS = C_ / 32;

  if (wid == 0) {  // prologue: stage 0 in flight
    tdm_load_2d(lds_off(&As[0][0][0]), aBase, C_, M_, 32, 128, C_);
    tdm_load_3d(lds_off(&Bs[0][0][0][0]), bBase, C_, H_, 3, 32, 64, 3,
                C_, (uint64_t)H_ * C_);
  }

  for (int kk = 0; kk < KSTEPS; ++kk) {
    const int cur = kk & 1, nxt = cur ^ 1;
    if (wid == 0) {
      if (kk + 1 < KSTEPS) {
        uint64_t koff = (uint64_t)(kk + 1) * 32 * 2;
        tdm_load_2d(lds_off(&As[nxt][0][0]), aBase + koff, C_, M_, 32, 128, C_);
        tdm_load_3d(lds_off(&Bs[nxt][0][0][0]), bBase + koff, C_, H_, 3, 32, 64, 3,
                    C_, (uint64_t)H_ * C_);
        __builtin_amdgcn_s_wait_tensorcnt(2);   // stage `cur` retired (in-order)
      } else {
        __builtin_amdgcn_s_wait_tensorcnt(0);
      }
    }
    __syncthreads();                            // stage `cur` visible to all waves

    union AV { v16bf v; v8bf h[2]; };
    AV a[2];
    #pragma unroll
    for (int i = 0; i < 2; ++i) {   // A frag: lane<16 K{0..7,16..23}; lane>=16 K{8..15,24..31}
      a[i].h[0] = *(const v8bf*)&As[cur][wm * 32 + i * 16 + fr][kh * 8];
      a[i].h[1] = *(const v8bf*)&As[cur][wm * 32 + i * 16 + fr][kh * 8 + 16];
    }
    #pragma unroll
    for (int r = 0; r < 3; ++r) {
      #pragma unroll
      for (int j = 0; j < 2; ++j) { // B frag: lane<16 K{0..7,16..23}; lane>=16 K{8..15,24..31}
        AV b;
        b.h[0] = *(const v8bf*)&Bs[cur][r][wn * 32 + j * 16 + fr][kh * 8];
        b.h[1] = *(const v8bf*)&Bs[cur][r][wn * 32 + j * 16 + fr][kh * 8 + 16];
        #pragma unroll
        for (int i = 0; i < 2; ++i)
          acc[r][i][j] = __builtin_amdgcn_wmma_f32_16x16x32_bf16(
              false, a[i].v, false, b.v, (short)0, acc[r][i][j], false, false);
      }
    }
    __syncthreads();                            // reads done before buffer reuse
  }

  // Epilogue: C layout -> lane: N = fr; VGPR v: M = v + 8*(lane>=16)
  const int g8 = (lane >> 4) * 8;
  #pragma unroll
  for (int i = 0; i < 2; ++i) {
    #pragma unroll
    for (int j = 0; j < 2; ++j) {
      int col = n0 + wn * 32 + j * 16 + fr;
      int rowBase = m0 + wm * 32 + i * 16 + g8;
      v8f a0 = acc[0][i][j], a1 = acc[1][i][j], a2 = acc[2][i][j];
      #pragma unroll
      for (int v = 0; v < 8; ++v) {
        size_t idx = (size_t)(rowBase + v) * H_ + col;
        float sv = a0[v];
        s_out[idx] = sv / (1.0f + __expf(-sv));   // silu
        float p1 = a1[v];
        poly_out[idx] = p1 + p1 * a2[v];          // c0 + c0*c1
      }
    }
  }
}

// ---------------- Phase 2: blocked causal scan ----------------

__global__ void k_scan_partial(const float* __restrict__ poly, float* __restrict__ csum) {
  int h = blockIdx.y * 256 + threadIdx.x;
  int bc = blockIdx.x;                 // b*NC_ + chunk
  int b = bc / NC_, c = bc % NC_;
  const float* p = poly + (size_t)(b * T_ + c * CT_) * H_ + h;
  float sum = 0.f;
  for (int t = 0; t < CT_; ++t) sum += p[(size_t)t * H_];
  csum[(size_t)bc * H_ + h] = sum;
}

__global__ void k_scan_offsets(float* __restrict__ csum) {
  int idx = blockIdx.x * 256 + threadIdx.x;   // over B*H
  int b = idx / H_, h = idx % H_;
  float run = 0.f;
  for (int c = 0; c < NC_; ++c) {
    size_t i = (size_t)(b * NC_ + c) * H_ + h;
    float v = csum[i];
    csum[i] = run;                             // exclusive prefix
    run += v;
  }
}

__global__ void k_scan_apply(const float* __restrict__ poly, const float* __restrict__ s,
                             const float* __restrict__ csum, bf16_t* __restrict__ yb) {
  int h = blockIdx.y * 256 + threadIdx.x;
  int bc = blockIdx.x;
  int b = bc / NC_, c = bc % NC_;
  float run = csum[(size_t)bc * H_ + h];
  size_t base = (size_t)(b * T_ + c * CT_) * H_ + h;
  for (int t = 0; t < CT_; ++t) {
    size_t i = base + (size_t)t * H_;
    run += poly[i];
    float agg = run / (float)(c * CT_ + t + 1);
    yb[i] = f2bf(s[i] * agg);                  // y = s * agg, bf16 for phase 3
  }
}

// ---------------- Phase 3: output WMMA GEMM (TDM double-buffered) ----------

__global__ __launch_bounds__(256)
void k_gemm_out(const bf16_t* __restrict__ yb,     // [M][H]
                const bf16_t* __restrict__ woutT,  // [C][H]
                float* __restrict__ out) {         // [M][C]
  __shared__ __align__(16) bf16_t As[2][128][32];
  __shared__ __align__(16) bf16_t Bs[2][64][32];

  const int tid = threadIdx.x, lane = tid & 31, wid = tid >> 5;
  const int wm = wid >> 1, wn = wid & 1;
  const int m0 = blockIdx.x * 128, n0 = blockIdx.y * 64;

  v8f acc[2][2] = {};
  const int fr = lane & 15, kh = lane >> 4;

  const uint64_t aBase = (uint64_t)(uintptr_t)yb    + (uint64_t)m0 * H_ * 2;
  const uint64_t bBase = (uint64_t)(uintptr_t)woutT + (uint64_t)n0 * H_ * 2;
  const int KSTEPS = H_ / 32;

  if (wid == 0) {
    tdm_load_2d(lds_off(&As[0][0][0]), aBase, H_, M_, 32, 128, H_);
    tdm_load_2d(lds_off(&Bs[0][0][0]), bBase, H_, C_, 32, 64, H_);
  }

  for (int kk = 0; kk < KSTEPS; ++kk) {
    const int cur = kk & 1, nxt = cur ^ 1;
    if (wid == 0) {
      if (kk + 1 < KSTEPS) {
        uint64_t koff = (uint64_t)(kk + 1) * 32 * 2;
        tdm_load_2d(lds_off(&As[nxt][0][0]), aBase + koff, H_, M_, 32, 128, H_);
        tdm_load_2d(lds_off(&Bs[nxt][0][0]), bBase + koff, H_, C_, 32, 64, H_);
        __builtin_amdgcn_s_wait_tensorcnt(2);
      } else {
        __builtin_amdgcn_s_wait_tensorcnt(0);
      }
    }
    __syncthreads();

    union AV { v16bf v; v8bf h[2]; };
    AV a[2];
    #pragma unroll
    for (int i = 0; i < 2; ++i) {
      a[i].h[0] = *(const v8bf*)&As[cur][wm * 32 + i * 16 + fr][kh * 8];
      a[i].h[1] = *(const v8bf*)&As[cur][wm * 32 + i * 16 + fr][kh * 8 + 16];
    }
    #pragma unroll
    for (int j = 0; j < 2; ++j) {
      AV b;
      b.h[0] = *(const v8bf*)&Bs[cur][wn * 32 + j * 16 + fr][kh * 8];
      b.h[1] = *(const v8bf*)&Bs[cur][wn * 32 + j * 16 + fr][kh * 8 + 16];
      #pragma unroll
      for (int i = 0; i < 2; ++i)
        acc[i][j] = __builtin_amdgcn_wmma_f32_16x16x32_bf16(
            false, a[i].v, false, b.v, (short)0, acc[i][j], false, false);
    }
    __syncthreads();
  }

  const int g8 = (lane >> 4) * 8;
  #pragma unroll
  for (int i = 0; i < 2; ++i)
    #pragma unroll
    for (int j = 0; j < 2; ++j) {
      int col = n0 + wn * 32 + j * 16 + fr;
      int rowBase = m0 + wm * 32 + i * 16 + g8;
      v8f a0 = acc[i][j];
      #pragma unroll
      for (int v = 0; v < 8; ++v)
        out[(size_t)(rowBase + v) * C_ + col] = a0[v];
    }
}

// ---------------- Launch ----------------

extern "C" void kernel_launch(void* const* d_in, const int* in_sizes, int n_in,
                              void* d_out, int out_size, void* d_ws, size_t ws_size,
                              hipStream_t stream) {
  (void)in_sizes; (void)n_in; (void)out_size; (void)ws_size;
  const float* x     = (const float*)d_in[0];
  const float* W_sel = (const float*)d_in[1];
  const float* W_agg = (const float*)d_in[2];
  const float* W_out = (const float*)d_in[3];
  const float* sqk_q = (const float*)d_in[4];
  const float* sqk_k = (const float*)d_in[5];
  float* out = (float*)d_out;

  uint8_t* ws = (uint8_t*)d_ws;
  size_t off = 0;
  bf16_t* xb    = (bf16_t*)(ws + off); off += (size_t)M_ * C_ * 2;        // 32 MB
  // wcat = [sel | agg-chunk0 | agg-chunk1], each [H][C] bf16, contiguous:
  bf16_t* wcat  = (bf16_t*)(ws + off);
  bf16_t* wselT = wcat;                off += (size_t)H_ * C_ * 2;        //  4 MB
  bf16_t* waggT = (bf16_t*)(ws + off); off += (size_t)2 * H_ * C_ * 2;    //  8 MB
  bf16_t* woutT = (bf16_t*)(ws + off); off += (size_t)C_ * H_ * 2;        //  4 MB
  float*  s_buf = (float*)(ws + off);  off += (size_t)M_ * H_ * 4;        // 128 MB
  float*  poly  = (float*)(ws + off);  off += (size_t)M_ * H_ * 4;        // 128 MB
  bf16_t* y_buf = (bf16_t*)(ws + off); off += (size_t)M_ * H_ * 2;        //  64 MB
  float*  csum  = (float*)(ws + off);  off += (size_t)B_ * NC_ * H_ * 4;  //   1 MB

  dim3 tb(32, 8);
  k_f32_to_bf16<<<4096, 256, 0, stream>>>(x, xb, M_ * C_);
  k_transpose_convert<<<dim3(C_ / 32, H_ / 32), tb, 0, stream>>>(W_sel, sqk_q, wselT, C_, H_);
  k_transpose_convert<<<dim3(C_ / 32, (2 * H_) / 32), tb, 0, stream>>>(W_agg, sqk_k, waggT, C_, 2 * H_);
  k_transpose_convert<<<dim3(H_ / 32, C_ / 32), tb, 0, stream>>>(W_out, nullptr, woutT, H_, C_);

  k_gemm_sel_agg<<<dim3(M_ / 128, H_ / 64), 256, 0, stream>>>(xb, wcat, s_buf, poly);

  k_scan_partial<<<dim3(B_ * NC_, H_ / 256), 256, 0, stream>>>(poly, csum);
  k_scan_offsets<<<(B_ * H_) / 256, 256, 0, stream>>>(csum);
  k_scan_apply<<<dim3(B_ * NC_, H_ / 256), 256, 0, stream>>>(poly, s_buf, csum, y_buf);

  k_gemm_out<<<dim3(M_ / 128, C_ / 64), 256, 0, stream>>>(y_buf, woutT, out);
}
